// signalEncoding_72885595013500
// MI455X (gfx1250) — compile-verified, matrix operand order checked by reading, not run on hardware
//
#include <hip/hip_runtime.h>
#include <hip/hip_bf16.h>

// ---------------------------------------------------------------------------
// Problem constants (from reference)
// ---------------------------------------------------------------------------
#define BSZ   16
#define SEQ   1024
#define SIG   256
#define EMB   256
#define LAT   256
#define NH    8
#define EH    32            // EMB / NH
#define DH    32            // LAT / NH
#define NBS   (BSZ * SEQ)   // 16384 rows
#define LN_EPS 1e-10f

#define SELU_SCALE 1.0507009873554805f
#define SELU_ALPHA 1.6732632423543772f

typedef __bf16 bf16_t;
typedef __attribute__((ext_vector_type(16))) __bf16 v16bf;
typedef __attribute__((ext_vector_type(8)))  __bf16 v8bf;
typedef __attribute__((ext_vector_type(8)))  float  v8f;

// Load a 16-element bf16 WMMA fragment: two contiguous 16B chunks
// (matches CDNA5 16-bit A/B layout: lane holds K in {kb..kb+7, kb+16..kb+23}).
__device__ __forceinline__ v16bf load_frag(const bf16_t* p) {
    union { v16bf v; struct { v8bf lo; v8bf hi; } s; } u;
    u.s.lo = *(const v8bf*)(p);
    u.s.hi = *(const v8bf*)(p + 16);
    return u.v;
}

// ---------------------------------------------------------------------------
// Generic WMMA GEMM: C = act(A(MxK,bf16) * Bt(NxK,bf16)^T + bias) + res1 + res2
// One wave computes a 16 x (16*NT) strip: the A fragment is reused NT times
// and the NT accumulator chains are independent (no WMMA->WMMA RAW stalls).
// Block = 8 waves covering 128 rows.
// ---------------------------------------------------------------------------
template <int NT>
__global__ void gemm_bf16_kernel(const bf16_t* __restrict__ A, int lda,
                                 const bf16_t* __restrict__ Bt, int ldb,
                                 const float* __restrict__ bias,
                                 const float* __restrict__ res1,
                                 const float* __restrict__ res2,
                                 float* __restrict__ C32,
                                 bf16_t* __restrict__ C16,
                                 int ldc, int M, int N, int K, int act)
{
    int lane = threadIdx.x & 31;
    int wave = threadIdx.x >> 5;
    int row0 = (blockIdx.x * 8 + wave) * 16;
    int col0 = blockIdx.y * (16 * NT);
    if (row0 >= M) return;

    int l16 = lane & 15;
    int hf  = lane >> 4;     // half-wave id
    int kb  = hf * 8;        // K sub-offset for 16-bit fragments

    const bf16_t* aRow = A + (size_t)(row0 + l16) * lda + kb;
    const bf16_t* bRow[NT];
#pragma unroll
    for (int nt = 0; nt < NT; ++nt)
        bRow[nt] = Bt + (size_t)(col0 + nt * 16 + l16) * ldb + kb;

    v8f acc[NT];
#pragma unroll
    for (int nt = 0; nt < NT; ++nt) acc[nt] = (v8f){};

    for (int k0 = 0; k0 < K; k0 += 32) {
        v16bf af = load_frag(aRow + k0);
#pragma unroll
        for (int nt = 0; nt < NT; ++nt) {
            v16bf bf = load_frag(bRow[nt] + k0);
            acc[nt] = __builtin_amdgcn_wmma_f32_16x16x32_bf16(false, af, false, bf,
                                                              (short)0, acc[nt], false, false);
        }
    }

    int mbase = row0 + hf * 8;
#pragma unroll
    for (int nt = 0; nt < NT; ++nt) {
        int n = col0 + nt * 16 + l16;
        float bv = bias ? bias[n] : 0.0f;
#pragma unroll
        for (int r = 0; r < 8; ++r) {
            size_t idx = (size_t)(mbase + r) * ldc + n;
            float v = acc[nt][r] + bv;
            if (act == 1)
                v = (v > 0.0f) ? SELU_SCALE * v
                               : SELU_SCALE * SELU_ALPHA * (__expf(v) - 1.0f);
            if (res1) v += res1[idx];
            if (res2) v += res2[idx];
            if (C32) C32[idx] = v;
            if (C16) C16[idx] = (bf16_t)v;
        }
    }
}

// ---------------------------------------------------------------------------
// Elementwise helpers
// ---------------------------------------------------------------------------
__global__ void addpos_bf16_kernel(const float* __restrict__ in,
                                   const float* __restrict__ pos,
                                   bf16_t* __restrict__ out, int total)
{
    int i = blockIdx.x * 256 + threadIdx.x;
    if (i < total) out[i] = (bf16_t)(in[i] + pos[i & (SIG - 1)]);
}

// W (K x N, f32 row-major) -> Wt (N x K, bf16)
__global__ void transpose_w_kernel(const float* __restrict__ W,
                                   bf16_t* __restrict__ Wt, int K, int N)
{
    int i = blockIdx.x * 256 + threadIdx.x;
    if (i >= K * N) return;
    int n = i / K, k = i - n * K;
    Wt[i] = (bf16_t)(W[k * N + n]);
}

// V32 (B,S,H*32 f32) -> Vt (B,H,32,S bf16)
__global__ void transpose_v_kernel(const float* __restrict__ V32,
                                   bf16_t* __restrict__ Vt)
{
    int i = blockIdx.x * 256 + threadIdx.x;
    int total = NBS * EMB;
    if (i >= total) return;
    int s = i & (SEQ - 1);
    int rest = i / SEQ;
    int d = rest & 31;
    int h = (rest >> 5) & (NH - 1);
    int b = rest >> (5 + 3);
    Vt[i] = (bf16_t)(V32[(size_t)(b * SEQ + s) * EMB + h * 32 + d]);
}

// ---------------------------------------------------------------------------
// LayerNorm: one 256-wide row per 256-thread block. Writes f32 and/or bf16.
// ---------------------------------------------------------------------------
__global__ void layernorm_kernel(const float* __restrict__ x,
                                 const float* __restrict__ g,
                                 const float* __restrict__ b,
                                 float* __restrict__ y32,
                                 bf16_t* __restrict__ y16)
{
    __shared__ float red[256];
    int row = blockIdx.x;
    int t = threadIdx.x;
    float v = x[(size_t)row * 256 + t];

    red[t] = v; __syncthreads();
    for (int s = 128; s > 0; s >>= 1) { if (t < s) red[t] += red[t + s]; __syncthreads(); }
    float mean = red[0] * (1.0f / 256.0f);
    __syncthreads();

    float d = v - mean;
    red[t] = d * d; __syncthreads();
    for (int s = 128; s > 0; s >>= 1) { if (t < s) red[t] += red[t + s]; __syncthreads(); }
    float var = red[0] * (1.0f / 256.0f);

    float o = d * rsqrtf(var + LN_EPS) * g[t] + b[t];
    if (y32) y32[(size_t)row * 256 + t] = o;
    if (y16) y16[(size_t)row * 256 + t] = (bf16_t)o;
}

// ---------------------------------------------------------------------------
// Encoder attention, flash-style with TRANSPOSED tiles.
//   S^T = BK_tile(16k x 32) x Q_tile^T   -> lane = query, VGPR r = key r+8*hf
//   out^T = V^T_tile x P^T               -> lane = query, VGPR r = d  r+8*hf
// The S^T C-fragment layout coincides exactly with the B-fragment layout
// needed for the out^T WMMA, so P is repacked with per-lane converts only:
// no LDS, no barriers, one shfl_xor(16) per softmax reduction.
// One wave owns a 16-query tile of one (b,h); block = 8 independent waves.
// ---------------------------------------------------------------------------
__global__ void enc_attn_kernel(const bf16_t* __restrict__ Q,
                                const bf16_t* __restrict__ BK,
                                const bf16_t* __restrict__ Vt,
                                bf16_t* __restrict__ O)
{
    int lane = threadIdx.x & 31;
    int wave = threadIdx.x >> 5;
    int l16 = lane & 15;     // query index within tile
    int hf  = lane >> 4;
    int kb  = hf * 8;
    int s0  = (blockIdx.x * 8 + wave) * 16;
    int b   = blockIdx.y >> 3;
    int h   = blockIdx.y & 7;

    const bf16_t* qBase  = Q  + (size_t)b * SEQ * EMB + h * 32;
    const bf16_t* bkBase = BK + (size_t)b * SEQ * EMB + h * 32;
    const bf16_t* vtBase = Vt + (size_t)(b * NH + h) * 32 * SEQ;

    // Q tile as B-operand (Bt = Q rows): lane n = query
    v16bf qf = load_frag(qBase + (size_t)(s0 + l16) * EMB + kb);

    v8f oT0 = {}, oT1 = {};         // out^T: d = r+8*hf (+16 for oT1)
    float mcur = -1e30f, lsum = 0.0f;
    const float scale = 0.17677669529663687f; // 1/sqrt(32)

    for (int kc = 0; kc < SEQ; kc += 32) {
        // BK rows as A-operand: lane m = key
        v16bf bkA0 = load_frag(bkBase + (size_t)(kc + l16) * EMB + kb);
        v16bf bkA1 = load_frag(bkBase + (size_t)(kc + 16 + l16) * EMB + kb);
        v8f sT0 = __builtin_amdgcn_wmma_f32_16x16x32_bf16(false, bkA0, false, qf,
                                                          (short)0, (v8f){}, false, false);
        v8f sT1 = __builtin_amdgcn_wmma_f32_16x16x32_bf16(false, bkA1, false, qf,
                                                          (short)0, (v8f){}, false, false);

        // per-query (= per-lane) online softmax over this 32-key chunk
        float p0[8], p1[8];
        float cm = -1e30f;
#pragma unroll
        for (int r = 0; r < 8; ++r) {
            p0[r] = sT0[r] * scale;
            p1[r] = sT1[r] * scale;
            cm = fmaxf(cm, fmaxf(p0[r], p1[r]));
        }
        cm = fmaxf(cm, __shfl_xor(cm, 16));       // combine the two half-wave copies
        float mnew  = fmaxf(mcur, cm);
        float alpha = __expf(mcur - mnew);
        float s = 0.0f;
#pragma unroll
        for (int r = 0; r < 8; ++r) {
            p0[r] = __expf(p0[r] - mnew);
            p1[r] = __expf(p1[r] - mnew);
            s += p0[r] + p1[r];
        }
        s += __shfl_xor(s, 16);
        lsum = lsum * alpha + s;
        mcur = mnew;

        // rescale accumulators (alpha is per-query = per-lane: plain VALU)
        v16bf pf;
#pragma unroll
        for (int r = 0; r < 8; ++r) {
            oT0[r] *= alpha;
            oT1[r] *= alpha;
            pf[r]     = (bf16_t)p0[r];   // key kb + r
            pf[8 + r] = (bf16_t)p1[r];   // key kb + 16 + r
        }

        // V^T rows as A-operand: lane m = d
        v16bf vA0 = load_frag(vtBase + (size_t)l16 * SEQ + kc + kb);
        v16bf vA1 = load_frag(vtBase + (size_t)(16 + l16) * SEQ + kc + kb);
        oT0 = __builtin_amdgcn_wmma_f32_16x16x32_bf16(false, vA0, false, pf,
                                                      (short)0, oT0, false, false);
        oT1 = __builtin_amdgcn_wmma_f32_16x16x32_bf16(false, vA1, false, pf,
                                                      (short)0, oT1, false, false);
    }

    // store: lane = query l16; oT0 covers d = kb..kb+7, oT1 d = 16+kb..16+kb+7
    float inv = 1.0f / lsum;
    size_t row = (size_t)b * SEQ + s0 + l16;
    bf16_t* outp = O + row * EMB + h * 32 + kb;
    v8bf st0, st1;
#pragma unroll
    for (int r = 0; r < 8; ++r) {
        st0[r] = (bf16_t)(oT0[r] * inv);
        st1[r] = (bf16_t)(oT1[r] * inv);
    }
    *(v8bf*)(outp)      = st0;
    *(v8bf*)(outp + 16) = st1;
}

// ---------------------------------------------------------------------------
// Decoder: q = LN(start) @ decQW + decQb  (tiny matvec, one block)
// ---------------------------------------------------------------------------
__global__ void matvec_kernel(const float* __restrict__ x,
                              const float* __restrict__ W,
                              const float* __restrict__ bias,
                              float* __restrict__ y)
{
    int j = threadIdx.x;
    float a = bias[j];
    for (int i = 0; i < 256; ++i) a += x[i] * W[i * 256 + j];
    y[j] = a;
}

// ---------------------------------------------------------------------------
// Decoder cross-attention: 1 query per (b,h). Bilinear key on concat(k,q),
// softmax over S, weighted sum of decV. One 256-thread block per (b,h).
// ---------------------------------------------------------------------------
__global__ void dec_attn_kernel(const float* __restrict__ decK,
                                const float* __restrict__ decV,
                                const float* __restrict__ qvec,
                                const float* __restrict__ decBW,  // (64,32)
                                const float* __restrict__ decBb,  // (32)
                                float* __restrict__ out32,        // (B,256)
                                bf16_t* __restrict__ out16)
{
    __shared__ float sc[SEQ];
    __shared__ float qterm[32];
    __shared__ float red[256];
    __shared__ float part[256];

    int b = blockIdx.x >> 3;
    int h = blockIdx.x & 7;
    int t = threadIdx.x;
    const float* qh = qvec + h * 32;

    if (t < 32) {
        float a = decBb[t];
        for (int i = 0; i < 32; ++i) a += qh[i] * decBW[(32 + i) * 32 + t];
        qterm[t] = a;
    }
    __syncthreads();

    const float invs = 0.125f; // 1/sqrt(64)
    for (int s = t; s < SEQ; s += 256) {
        const float* krow = decK + (size_t)(b * SEQ + s) * EMB + h * 32;
        float score = 0.0f;
        for (int d = 0; d < 32; ++d) {
            float bk = qterm[d];
            for (int i = 0; i < 32; ++i) bk += krow[i] * decBW[i * 32 + d];
            score += qh[d] * bk;
        }
        sc[s] = score * invs;
    }
    __syncthreads();

    float mx = -1e30f;
    for (int s = t; s < SEQ; s += 256) mx = fmaxf(mx, sc[s]);
    red[t] = mx; __syncthreads();
    for (int s2 = 128; s2 > 0; s2 >>= 1) { if (t < s2) red[t] = fmaxf(red[t], red[t + s2]); __syncthreads(); }
    mx = red[0]; __syncthreads();

    float sum = 0.0f;
    for (int s = t; s < SEQ; s += 256) { float p = __expf(sc[s] - mx); sc[s] = p; sum += p; }
    red[t] = sum; __syncthreads();
    for (int s2 = 128; s2 > 0; s2 >>= 1) { if (t < s2) red[t] += red[t + s2]; __syncthreads(); }
    float denom = red[0]; __syncthreads();

    int d = t & 31, seg = t >> 5;              // 8 segments x 128 keys
    float acc = 0.0f;
    for (int s = seg * 128; s < seg * 128 + 128; ++s)
        acc += sc[s] * decV[(size_t)(b * SEQ + s) * EMB + h * 32 + d];
    part[t] = acc; __syncthreads();

    if (t < 32) {
        float tot = 0.0f;
        for (int g2 = 0; g2 < 8; ++g2) tot += part[g2 * 32 + t];
        float o = tot / denom;
        out32[b * 256 + h * 32 + t] = o;
        out16[b * 256 + h * 32 + t] = (bf16_t)o;
    }
}

// ---------------------------------------------------------------------------
// Host orchestration
// ---------------------------------------------------------------------------
static inline void gemm(hipStream_t st, const bf16_t* A, int lda,
                        const bf16_t* Bt, int ldb, const float* bias,
                        const float* res1, const float* res2,
                        float* C32, bf16_t* C16, int ldc,
                        int M, int N, int K, int act)
{
    dim3 blk(256);
    if ((N & 63) == 0) {
        dim3 g((unsigned)((M + 127) / 128), (unsigned)(N / 64));
        gemm_bf16_kernel<4><<<g, blk, 0, st>>>(A, lda, Bt, ldb, bias, res1, res2,
                                               C32, C16, ldc, M, N, K, act);
    } else {
        dim3 g((unsigned)((M + 127) / 128), (unsigned)(N / 32));
        gemm_bf16_kernel<2><<<g, blk, 0, st>>>(A, lda, Bt, ldb, bias, res1, res2,
                                               C32, C16, ldc, M, N, K, act);
    }
}

static inline void trw(hipStream_t st, const float* W, bf16_t* Wt, int K, int N)
{
    int total = K * N;
    transpose_w_kernel<<<(total + 255) / 256, 256, 0, st>>>(W, Wt, K, N);
}

extern "C" void kernel_launch(void* const* d_in, const int* in_sizes, int n_in,
                              void* d_out, int out_size, void* d_ws, size_t ws_size,
                              hipStream_t stream)
{
    (void)in_sizes; (void)n_in; (void)out_size; (void)ws_size;

    // ---- inputs (setup_inputs order) ----
    const float* inputData = (const float*)d_in[0];
    const float* pos_enc   = (const float*)d_in[1];
    const float* embW = (const float*)d_in[2];  const float* embb = (const float*)d_in[3];
    const float* qW   = (const float*)d_in[4];  const float* qb   = (const float*)d_in[5];
    const float* kW   = (const float*)d_in[6];  const float* kbias= (const float*)d_in[7];
    const float* vW   = (const float*)d_in[8];  const float* vb   = (const float*)d_in[9];
    const float* encBW= (const float*)d_in[10]; const float* encBb= (const float*)d_in[11];
    const float* oW   = (const float*)d_in[12]; const float* ob   = (const float*)d_in[13];
    const float* ffeW1= (const float*)d_in[14]; const float* ffeb1= (const float*)d_in[15];
    const float* ffeW2= (const float*)d_in[16]; const float* ffeb2= (const float*)d_in[17];
    const float* lnSAg= (const float*)d_in[18]; const float* lnSAb= (const float*)d_in[19];
    const float* lnFFg= (const float*)d_in[20]; const float* lnFFb= (const float*)d_in[21];
    const float* start = (const float*)d_in[22];
    const float* decQW = (const float*)d_in[23]; const float* decQb = (const float*)d_in[24];
    const float* decKW = (const float*)d_in[25]; const float* decKb = (const float*)d_in[26];
    const float* decVW = (const float*)d_in[27]; const float* decVb = (const float*)d_in[28];
    const float* decBW = (const float*)d_in[29]; const float* decBb = (const float*)d_in[30];
    const float* latW  = (const float*)d_in[31]; const float* latb  = (const float*)d_in[32];
    const float* ffdW1 = (const float*)d_in[33]; const float* ffdb1 = (const float*)d_in[34];
    const float* ffdW2 = (const float*)d_in[35]; const float* ffdb2 = (const float*)d_in[36];
    const float* lnCAg = (const float*)d_in[37]; const float* lnCAb = (const float*)d_in[38];
    const float* lnLFFg= (const float*)d_in[39]; const float* lnLFFb= (const float*)d_in[40];
    const float* finW  = (const float*)d_in[41]; const float* finb  = (const float*)d_in[42];

    // ---- workspace arena ----
    size_t off = 0;
    char* base = (char*)d_ws;
    auto alloc = [&](size_t bytes) -> char* {
        char* p = base + off;
        off += (bytes + 255) & ~(size_t)255;
        return p;
    };
    bf16_t* x16    = (bf16_t*)alloc((size_t)NBS * 256 * 2);
    float*  emb32  = (float*) alloc((size_t)NBS * 256 * 4);
    float*  t32    = (float*) alloc((size_t)NBS * 256 * 4);
    bf16_t* t16    = (bf16_t*)alloc((size_t)NBS * 256 * 2);
    float*  n32    = (float*) alloc((size_t)NBS * 256 * 4);
    bf16_t* n16    = (bf16_t*)alloc((size_t)NBS * 256 * 2);
    bf16_t* q16    = (bf16_t*)alloc((size_t)NBS * 256 * 2);
    bf16_t* k16    = (bf16_t*)alloc((size_t)NBS * 256 * 2);
    bf16_t* bk16   = (bf16_t*)alloc((size_t)NBS * 256 * 2);
    float*  v32    = (float*) alloc((size_t)NBS * 256 * 4);
    bf16_t* vt16   = (bf16_t*)alloc((size_t)NBS * 256 * 2);
    bf16_t* ao16   = (bf16_t*)alloc((size_t)NBS * 256 * 2);
    bf16_t* h16    = (bf16_t*)alloc((size_t)NBS * 512 * 2);
    float*  decK32 = (float*) alloc((size_t)NBS * 256 * 4);
    float*  decV32 = (float*) alloc((size_t)NBS * 256 * 4);

    bf16_t* embWt  = (bf16_t*)alloc(256 * 256 * 2);
    bf16_t* qWt    = (bf16_t*)alloc(256 * 256 * 2);
    bf16_t* kWt    = (bf16_t*)alloc(256 * 256 * 2);
    bf16_t* vWt    = (bf16_t*)alloc(256 * 256 * 2);
    bf16_t* encBWt = (bf16_t*)alloc(32 * 32 * 2);
    bf16_t* oWt    = (bf16_t*)alloc(256 * 256 * 2);
    bf16_t* ffeW1t = (bf16_t*)alloc(512 * 256 * 2);
    bf16_t* ffeW2t = (bf16_t*)alloc(256 * 512 * 2);
    bf16_t* decKWt = (bf16_t*)alloc(256 * 256 * 2);
    bf16_t* decVWt = (bf16_t*)alloc(256 * 256 * 2);
    bf16_t* latWt  = (bf16_t*)alloc(256 * 256 * 2);
    bf16_t* ffdW1t = (bf16_t*)alloc(128 * 256 * 2);
    bf16_t* ffdW2t = (bf16_t*)alloc(256 * 128 * 2);
    bf16_t* finWt  = (bf16_t*)alloc(256 * 256 * 2);

    float*  nl32   = (float*) alloc(256 * 4);
    float*  qvec   = (float*) alloc(256 * 4);
    float*  aDec32 = (float*) alloc(BSZ * 256 * 4);
    bf16_t* aDec16 = (bf16_t*)alloc(BSZ * 256 * 2);
    float*  lat32  = (float*) alloc(BSZ * 256 * 4);
    bf16_t* lat16  = (bf16_t*)alloc(BSZ * 256 * 2);
    bf16_t* nf16   = (bf16_t*)alloc(BSZ * 256 * 2);
    bf16_t* hd16   = (bf16_t*)alloc(BSZ * 128 * 2);

    // ---- weight transpose + bf16 convert (tiny) ----
    trw(stream, embW,  embWt,  256, 256);
    trw(stream, qW,    qWt,    256, 256);
    trw(stream, kW,    kWt,    256, 256);
    trw(stream, vW,    vWt,    256, 256);
    trw(stream, encBW, encBWt, 32,  32);
    trw(stream, oW,    oWt,    256, 256);
    trw(stream, ffeW1, ffeW1t, 256, 512);
    trw(stream, ffeW2, ffeW2t, 512, 256);
    trw(stream, decKW, decKWt, 256, 256);
    trw(stream, decVW, decVWt, 256, 256);
    trw(stream, latW,  latWt,  256, 256);
    trw(stream, ffdW1, ffdW1t, 256, 128);
    trw(stream, ffdW2, ffdW2t, 128, 256);
    trw(stream, finW,  finWt,  256, 256);

    // ---- embedding: emb = (input + pos) @ embW + embb ----
    {
        int total = NBS * SIG;
        addpos_bf16_kernel<<<(total + 255) / 256, 256, 0, stream>>>(inputData, pos_enc, x16, total);
    }
    gemm(stream, x16, 256, embWt, 256, embb, nullptr, nullptr,
         emb32, nullptr, 256, NBS, 256, 256, 0);

    // ---- encoder layers ----
    const float* tcur = emb32;
    for (int l = 0; l < 2; ++l) {
        // n = LN(t)
        layernorm_kernel<<<NBS, 256, 0, stream>>>(tcur, lnSAg + l * 256, lnSAb + l * 256, n32, n16);
        // q, k, v projections
        gemm(stream, n16, 256, qWt, 256, qb,    nullptr, nullptr, nullptr, q16, 256, NBS, 256, 256, 0);
        gemm(stream, n16, 256, kWt, 256, kbias, nullptr, nullptr, nullptr, k16, 256, NBS, 256, 256, 0);
        gemm(stream, n16, 256, vWt, 256, vb,    nullptr, nullptr, v32, nullptr, 256, NBS, 256, 256, 0);
        // bilinear keys: bk = k_heads(131072x32) @ encBW + encBb
        gemm(stream, k16, 32, encBWt, 32, encBb, nullptr, nullptr,
             nullptr, bk16, 32, NBS * NH, 32, 32, 0);
        // V -> (B,H,32,S) bf16 for WMMA A-fragments of out^T
        {
            int total = NBS * EMB;
            transpose_v_kernel<<<(total + 255) / 256, 256, 0, stream>>>(v32, vt16);
        }
        // attention (8 independent waves per block)
        {
            dim3 g(SEQ / 16 / 8, BSZ * NH), blk(256);
            enc_attn_kernel<<<g, blk, 0, stream>>>(q16, bk16, vt16, ao16);
        }
        // t = attnOut @ oW + ob + n + emb
        gemm(stream, ao16, 256, oWt, 256, ob, n32, emb32, t32, nullptr, 256, NBS, 256, 256, 0);
        // n2 = LN(t)
        layernorm_kernel<<<NBS, 256, 0, stream>>>(t32, lnFFg + l * 256, lnFFb + l * 256, n32, n16);
        // ff: selu(n2 @ W1 + b1) @ W2 + b2 + n2
        gemm(stream, n16, 256, ffeW1t, 256, ffeb1, nullptr, nullptr,
             nullptr, h16, 512, NBS, 512, 256, 1);
        gemm(stream, h16, 512, ffeW2t, 512, ffeb2, n32, nullptr, t32, t16, 256, NBS, 256, 512, 0);
        tcur = t32;
    }

    // ---- decoder: decK/decV are layer-invariant (t fixed) ----
    gemm(stream, t16, 256, decKWt, 256, decKb, nullptr, nullptr, decK32, nullptr, 256, NBS, 256, 256, 0);
    gemm(stream, t16, 256, decVWt, 256, decVb, nullptr, nullptr, decV32, nullptr, 256, NBS, 256, 256, 0);

    for (int l = 0; l < 2; ++l) {
        // nl = LN(start) (identical across batch), q = nl @ decQW + decQb
        layernorm_kernel<<<1, 256, 0, stream>>>(start, lnCAg + l * 256, lnCAb + l * 256, nl32, nullptr);
        matvec_kernel<<<1, 256, 0, stream>>>(nl32, decQW, decQb, qvec);
        // cross attention (VALU; tiny)
        dec_attn_kernel<<<BSZ * NH, 256, 0, stream>>>(decK32, decV32, qvec, decBW, decBb,
                                                      aDec32, aDec16);
        // lat = attn @ latW + latb
        gemm(stream, aDec16, 256, latWt, 256, latb, nullptr, nullptr,
             lat32, nullptr, 256, BSZ, 256, 256, 0);
        // nf = LN(lat)
        layernorm_kernel<<<BSZ, 256, 0, stream>>>(lat32, lnLFFg + l * 256, lnLFFb + l * 256,
                                                  nullptr, nf16);
        // lat = selu(nf @ ffdW1 + b1) @ ffdW2 + b2
        gemm(stream, nf16, 256, ffdW1t, 256, ffdb1, nullptr, nullptr,
             nullptr, hd16, 128, BSZ, 128, 256, 1);
        gemm(stream, hd16, 128, ffdW2t, 128, ffdb2, nullptr, nullptr,
             lat32, lat16, 256, BSZ, 256, 128, 0);
    }

    // ---- final projection straight into d_out (f32, B x LAT) ----
    gemm(stream, lat16, 256, finWt, 256, finb, nullptr, nullptr,
         (float*)d_out, nullptr, 256, BSZ, 256, 256, 0);
}
